// DotProductAttention_31748398252253
// MI455X (gfx1250) — matrix operand order repaired and checked by reading, not compiled, not run on hardware
//
#include <hip/hip_runtime.h>
#include <hip/hip_bf16.h>
#include <math.h>

typedef _Float16 f16;
typedef __attribute__((ext_vector_type(16))) _Float16 v16h;
typedef __attribute__((ext_vector_type(8)))  _Float16 v8h;
typedef __attribute__((ext_vector_type(4)))  _Float16 v4h;
typedef __attribute__((ext_vector_type(8)))  float    v8f;
typedef __attribute__((ext_vector_type(4)))  float    v4f;

#define B_   32
#define LQ_  2048
#define LK_  2048
#define D_   128
#define BM   128          // query rows per workgroup (8 waves x 16 rows)
#define BN   64           // keys per iteration
#define NW   8
#define KSS  (D_ + 8)     // K-tile LDS row stride (f16) -- pad vs bank conflicts
#define VTS  (BN + 8)     // transposed-V LDS row stride (f16)
#define PS   (BN + 8)     // P-buffer row stride (f16)

// Load a 16x16 B-fragment row (16 contiguous f16) as two aligned b128s.
__device__ __forceinline__ v16h ld_b16(const f16* p) {
  v8h lo = *(const v8h*)p;
  v8h hi = *(const v8h*)(p + 8);
  return __builtin_shufflevector(lo, hi, 0,1,2,3,4,5,6,7,8,9,10,11,12,13,14,15);
}

__global__ __launch_bounds__(256, 1)
void DotProductAttention_fa_kernel(const float* __restrict__ Qg,
                                   const float* __restrict__ Kg,
                                   const float* __restrict__ Vg,
                                   const int*   __restrict__ VLg,
                                   float*       __restrict__ Og) {
  __shared__ f16 Ks_s[BN * KSS];      // K tile, row-major [key][d]
  __shared__ f16 Vt_s[D_ * VTS];      // V tile, transposed [d][key]
  __shared__ f16 P_s[NW * 16 * PS];   // per-wave probability buffer

  const int tid  = threadIdx.x;
  const int wave = tid >> 5;
  const int lane = tid & 31;
  const int half = lane >> 4;
  const int lp   = lane & 15;
  const int b    = blockIdx.y;
  const int q0   = blockIdx.x * BM;

  const int valid = VLg[b];
  const int nkb   = (valid + BN - 1) / BN;

  // ---- Q fragments in WMMA A-layout, scale folded into f16 convert ----
  const float scale = 0.08838834764831845f; // 1/sqrt(128)
  const int qrow = q0 + wave * 16 + lp;
  const float* Qrow = Qg + ((size_t)b * LQ_ + qrow) * D_;
  v16h qa[4];
  #pragma unroll
  for (int ks = 0; ks < 4; ++ks) {
    const int d0 = ks * 32 + 8 * half;
    v4f a0 = *(const v4f*)(Qrow + d0);
    v4f a1 = *(const v4f*)(Qrow + d0 + 4);
    v4f a2 = *(const v4f*)(Qrow + d0 + 16);
    v4f a3 = *(const v4f*)(Qrow + d0 + 20);
    #pragma unroll
    for (int j = 0; j < 4; ++j) {
      qa[ks][j]      = (f16)(a0[j] * scale);
      qa[ks][4 + j]  = (f16)(a1[j] * scale);
      qa[ks][8 + j]  = (f16)(a2[j] * scale);
      qa[ks][12 + j] = (f16)(a3[j] * scale);
    }
  }

  // ---- prefetch first K/V tile into WGP cache (256 x 128B = 32KB each) ----
  {
    const float* pk = Kg + ((size_t)b * LK_) * D_;
    const float* pv = Vg + ((size_t)b * LK_) * D_;
    __builtin_prefetch(pk + tid * 32, 0, 3);
    __builtin_prefetch(pv + tid * 32, 0, 3);
  }

  // ---- accumulators + online-softmax state (rows r+8*half) ----
  v8f O[8];
  #pragma unroll
  for (int t = 0; t < 8; ++t)
    #pragma unroll
    for (int r = 0; r < 8; ++r) O[t][r] = 0.f;
  float m_r[8], l_r[8];
  #pragma unroll
  for (int r = 0; r < 8; ++r) { m_r[r] = -1e30f; l_r[r] = 0.f; }

  f16* Pw = &P_s[wave * 16 * PS];

  for (int kb = 0; kb < nkb; ++kb) {
    __syncthreads();  // previous tile fully consumed
    // ---- cooperative staging: K row-major f16, V transposed f16 ----
    {
      const float* sk = Kg + ((size_t)b * LK_ + (size_t)kb * BN) * D_;
      const float* sv = Vg + ((size_t)b * LK_ + (size_t)kb * BN) * D_;
      #pragma unroll
      for (int it = 0; it < (BN * D_ / 4) / 256; ++it) {
        int i   = tid + it * 256;
        int row = i >> 5;              // D_/4 == 32 float4 per row
        int c4  = (i & 31) * 4;
        v4f vk = *(const v4f*)(sk + row * D_ + c4);
        v4h hk;
        hk[0] = (f16)vk[0]; hk[1] = (f16)vk[1];
        hk[2] = (f16)vk[2]; hk[3] = (f16)vk[3];
        *(v4h*)(&Ks_s[row * KSS + c4]) = hk;

        v4f vv = *(const v4f*)(sv + row * D_ + c4);
        #pragma unroll
        for (int j = 0; j < 4; ++j)
          Vt_s[(c4 + j) * VTS + row] = (f16)vv[j];
      }
      // prefetch next tile while this one is consumed (no counters, no regs)
      if (kb + 1 < nkb) {
        const float* pk = sk + (size_t)BN * D_;
        const float* pv = sv + (size_t)BN * D_;
        __builtin_prefetch(pk + tid * 32, 0, 3);
        __builtin_prefetch(pv + tid * 32, 0, 3);
      }
    }
    __syncthreads();

    // ---- S = Q @ K^T : 4 key tiles x 4 D-steps of WMMA ----
    v8f S[4];
    #pragma unroll
    for (int nt = 0; nt < 4; ++nt) {
      v8f acc;
      #pragma unroll
      for (int r = 0; r < 8; ++r) acc[r] = 0.f;
      #pragma unroll
      for (int ks = 0; ks < 4; ++ks) {
        const f16* kp = &Ks_s[(nt * 16 + lp) * KSS + ks * 32 + 8 * half];
        v16h bf = __builtin_shufflevector(*(const v8h*)kp, *(const v8h*)(kp + 16),
                    0,1,2,3,4,5,6,7,8,9,10,11,12,13,14,15);
        acc = __builtin_amdgcn_wmma_f32_16x16x32_f16(false, qa[ks], false, bf,
                                                     (short)0, acc, false, false);
      }
      S[nt] = acc;
    }

    // ---- mask invalid keys (column index is lane-determined) ----
    #pragma unroll
    for (int nt = 0; nt < 4; ++nt) {
      int n = kb * BN + nt * 16 + lp;
      if (n >= valid) {
        #pragma unroll
        for (int r = 0; r < 8; ++r) S[nt][r] = -1e30f;
      }
    }

    // ---- online softmax: per owned row (row = r + 8*half) ----
    #pragma unroll
    for (int r = 0; r < 8; ++r) {
      float mx = S[0][r];
      #pragma unroll
      for (int nt = 1; nt < 4; ++nt) mx = fmaxf(mx, S[nt][r]);
      #pragma unroll
      for (int off = 8; off >= 1; off >>= 1)
        mx = fmaxf(mx, __shfl_xor(mx, off, 32));
      float mnew  = fmaxf(m_r[r], mx);
      float alpha = __expf(m_r[r] - mnew);
      float rs = 0.f;
      #pragma unroll
      for (int nt = 0; nt < 4; ++nt) {
        float p = __expf(S[nt][r] - mnew);
        S[nt][r] = p;
        rs += p;
      }
      #pragma unroll
      for (int off = 8; off >= 1; off >>= 1)
        rs += __shfl_xor(rs, off, 32);
      l_r[r] = l_r[r] * alpha + rs;
      m_r[r] = mnew;
      #pragma unroll
      for (int t = 0; t < 8; ++t) O[t][r] *= alpha;
    }

    // ---- C-layout -> A-layout via per-wave LDS round trip ----
    #pragma unroll
    for (int nt = 0; nt < 4; ++nt)
      #pragma unroll
      for (int r = 0; r < 8; ++r)
        Pw[(r + 8 * half) * PS + nt * 16 + lp] = (f16)S[nt][r];
    // same-wave LDS ops are in-order: store->load hazard-free

    // ---- O += P @ V : 2 key-steps x 8 D-tiles of WMMA ----
    #pragma unroll
    for (int ks2 = 0; ks2 < 2; ++ks2) {
      const f16* rp = Pw + lp * PS + ks2 * 32 + 8 * half;
      v8h c0 = *(const v8h*)rp;         // A elems 0..7  (K = base+0..7)
      v8h c1 = *(const v8h*)(rp + 16);  // A elems 8..15 (K = base+16..23)
      v16h pa = __builtin_shufflevector(c0, c1,
                  0,1,2,3,4,5,6,7,8,9,10,11,12,13,14,15);
      #pragma unroll
      for (int t = 0; t < 8; ++t) {
        const f16* vp = &Vt_s[(t * 16 + lp) * VTS + ks2 * 32 + 8 * half];
        v16h bv = __builtin_shufflevector(*(const v8h*)vp, *(const v8h*)(vp + 16),
                    0,1,2,3,4,5,6,7,8,9,10,11,12,13,14,15);
        O[t] = __builtin_amdgcn_wmma_f32_16x16x32_f16(false, pa, false, bv,
                                                      (short)0, O[t], false, false);
      }
    }
  }

  // ---- epilogue: normalize by l, store fp32 ----
  float* Orow = Og + ((size_t)b * LQ_ + q0 + wave * 16) * D_;
  #pragma unroll
  for (int r = 0; r < 8; ++r) {
    float inv = 1.0f / l_r[r];
    int row = r + 8 * half;
    #pragma unroll
    for (int t = 0; t < 8; ++t)
      Orow[row * D_ + t * 16 + lp] = O[t][r] * inv;
  }
}

extern "C" void kernel_launch(void* const* d_in, const int* in_sizes, int n_in,
                              void* d_out, int out_size, void* d_ws, size_t ws_size,
                              hipStream_t stream) {
  (void)in_sizes; (void)n_in; (void)out_size; (void)d_ws; (void)ws_size;
  const float* Q  = (const float*)d_in[0];
  const float* K  = (const float*)d_in[1];
  const float* V  = (const float*)d_in[2];
  const int*   VL = (const int*)d_in[3];
  float* out = (float*)d_out;

  dim3 grid(LQ_ / BM, B_);
  dim3 block(256);
  DotProductAttention_fa_kernel<<<grid, block, 0, stream>>>(Q, K, V, VL, out);
}
